// DynamicAttention_61211873903413
// MI455X (gfx1250) — matrix operand order repaired and checked by reading, not compiled, not run on hardware
//
#include <hip/hip_runtime.h>
#include <hip/hip_bf16.h>
#include <stdint.h>

// ---------------------------------------------------------------------------
// CDNA5 / gfx1250 fused attention block:
//   bf16 WMMA (16x16x32, f32 accum) + async global->LDS staging (ASYNCcnt).
// ---------------------------------------------------------------------------

typedef __bf16 bf16_t;
typedef bf16_t v16bf __attribute__((ext_vector_type(16)));
typedef float  v8f   __attribute__((ext_vector_type(8)));

union Frag { v16bf v; unsigned int u[8]; };

__device__ __forceinline__ unsigned short f2bf(float f) {
    unsigned int u = __float_as_uint(f);
    u += 0x7FFFu + ((u >> 16) & 1u);          // round-to-nearest-even
    return (unsigned short)(u >> 16);
}

__device__ __forceinline__ v8f wmma_bf16(const Frag& a, const Frag& b, v8f c) {
    return __builtin_amdgcn_wmma_f32_16x16x32_bf16(
        false, a.v, false, b.v, (short)0, c, false, false);
}

// k index inside a 16-bit A fragment (16x32): lane half h16, vgpr j
__device__ __forceinline__ int a_kidx(int j, int h16) {
    return (j < 4) ? (8 * h16 + 2 * j) : (16 + 8 * h16 + 2 * (j - 4));
}
// k index inside a 16-bit B fragment (32x16): lane half h16, vgpr j
__device__ __forceinline__ int b_kidx(int j, int h16) {
    return (j < 4) ? (8 * h16 + 2 * j) : (16 + 8 * h16 + 2 * (j - 4));
}

// LDS byte offset = low 32 bits of the flat address (LDS aperture keeps the
// offset in addr[31:0], ISA 10.2).
__device__ __forceinline__ unsigned lds_off(const void* p) {
    return (unsigned)(uintptr_t)p;
}

// Async global -> LDS copies (VGLOBAL GLOBAL_LOAD_ASYNC_TO_LDS_*, ASYNCcnt).
__device__ __forceinline__ void async_ld_b64(unsigned lds, const void* g) {
    asm volatile("global_load_async_to_lds_b64 %0, %1, off"
                 :: "v"(lds), "v"(g) : "memory");
}
__device__ __forceinline__ void async_ld_b128(unsigned lds, const void* g) {
    asm volatile("global_load_async_to_lds_b128 %0, %1, off"
                 :: "v"(lds), "v"(g) : "memory");
}
__device__ __forceinline__ void wait_async_0() {
    asm volatile("s_wait_asynccnt 0x0" ::: "memory");
}
__device__ __forceinline__ void wait_async_2() {
    asm volatile("s_wait_asynccnt 0x2" ::: "memory");
}

#define EPS 1e-5f

// ---------------------------------------------------------------------------
// Kernel 1: QKV projection GEMM  out[o][n] = BN(sum_c W[o][c] * x[b][c][n])
//   C = 384, N = 1024.  Workgroup tile 128(M) x 64(N), BK = 32, 8 waves.
//   mode 0: write bf16 transposed per head ->  out[b][h][n][32]   (Q, K)
//   mode 1: write bf16 natural             ->  out[b][o][n]       (V)
//   (both operands converted f32->bf16 in-flight, so fills stay manual)
// ---------------------------------------------------------------------------
__global__ __launch_bounds__(256) void gemm_qkv_kernel(
    const float* __restrict__ W, const float* __restrict__ X,
    const float* __restrict__ bng, const float* __restrict__ bnb,
    const float* __restrict__ bnm, const float* __restrict__ bnv,
    unsigned short* __restrict__ out, int M, int mode)
{
    const int C = 384, N = 1024;
    __shared__ __align__(16) unsigned short As[128][34];  // [m][k]
    __shared__ __align__(16) unsigned short Bs[64][34];   // [n][k]

    const int b  = blockIdx.z;
    const int o0 = blockIdx.y * 128;
    const int n0 = blockIdx.x * 64;
    const int t  = threadIdx.x;
    const int wv = t >> 5, lane = t & 31;
    const int m16 = lane & 15, h16 = lane >> 4;
    const int om = (wv >> 1) * 32, on = (wv & 1) * 32;

    const float* Xb = X + (size_t)b * C * N;

    v8f acc[2][2];
    for (int i = 0; i < 2; ++i)
        for (int j = 0; j < 2; ++j) acc[i][j] = {};

    for (int kb = 0; kb < C; kb += 32) {
        #pragma unroll
        for (int i = 0; i < 16; ++i) {
            int idx = t + i * 256, r = idx >> 5, k = idx & 31;
            As[r][k] = f2bf(W[(size_t)(o0 + r) * C + kb + k]);
        }
        #pragma unroll
        for (int i = 0; i < 8; ++i) {
            int idx = t + i * 256, n = idx & 63, c = idx >> 6;
            Bs[n][c] = f2bf(Xb[(size_t)(kb + c) * N + n0 + n]);
        }
        __syncthreads();

        Frag a0, a1, bf0, bf1;
        #pragma unroll
        for (int j = 0; j < 8; ++j) {
            int ka = a_kidx(j, h16);
            int kbb = b_kidx(j, h16);
            a0.u[j]  = *(const unsigned int*)&As[om + m16][ka];
            a1.u[j]  = *(const unsigned int*)&As[om + 16 + m16][ka];
            bf0.u[j] = *(const unsigned int*)&Bs[on + m16][kbb];
            bf1.u[j] = *(const unsigned int*)&Bs[on + 16 + m16][kbb];
        }
        acc[0][0] = wmma_bf16(a0, bf0, acc[0][0]);
        acc[0][1] = wmma_bf16(a0, bf1, acc[0][1]);
        acc[1][0] = wmma_bf16(a1, bf0, acc[1][0]);
        acc[1][1] = wmma_bf16(a1, bf1, acc[1][1]);
        __syncthreads();
    }

    #pragma unroll
    for (int ti = 0; ti < 2; ++ti)
        #pragma unroll
        for (int tj = 0; tj < 2; ++tj)
            #pragma unroll
            for (int j = 0; j < 8; ++j) {
                int o = o0 + om + ti * 16 + j + 8 * h16;
                int n = n0 + on + tj * 16 + m16;
                float sc = bng[o] * rsqrtf(bnv[o] + EPS);
                float sh = bnb[o] - bnm[o] * sc;
                unsigned short r = f2bf(acc[ti][tj][j] * sc + sh);
                if (mode == 0) {   // [b][h][n][32]
                    int h = o >> 5, d = o & 31;
                    out[(((size_t)b * (M >> 5) + h) * 1024 + n) * 32 + d] = r;
                } else {           // [b][o][n]
                    out[((size_t)b * M + o) * (size_t)N + n] = r;
                }
            }
}

// ---------------------------------------------------------------------------
// Kernel 2: flash attention per (b,h).  Grid: (1024/128, B*H).
//   qt, kt: [b][h][n][32] bf16      vv: [b][h][e][n] bf16
//   at out: [b][n][512]   bf16 (relu applied)
//   K/V tiles are staged workgroup-wide through LDS with double-buffered
//   async copies (tile i+1 in flight while tile i is consumed by WMMA).
// ---------------------------------------------------------------------------
__global__ __launch_bounds__(256) void attn_kernel(
    const unsigned short* __restrict__ qt,
    const unsigned short* __restrict__ kt,
    const unsigned short* __restrict__ vv,
    unsigned short* __restrict__ at)
{
    const int bh = blockIdx.y;            // 0..63
    const int b = bh >> 3, h = bh & 7;
    const int t = threadIdx.x;
    const int wv = t >> 5, lane = t & 31;
    const int m16 = lane & 15, h16 = lane >> 4;
    const int n0 = blockIdx.x * 128 + wv * 16;

    // 80B row pitch: 16B-aligned for B128, banks (20*r mod 64) conflict-free
    __shared__ __align__(16) unsigned short Ks[2][32][40];  // [buf][key][d]
    __shared__ __align__(16) unsigned short Vs[2][64][40];  // [buf][e][m]
    __shared__ __align__(16) unsigned short Pl[8][16][34];  // per-wave P patch

    const unsigned short* qb = qt + (size_t)bh * 1024 * 32;
    const unsigned short* kb = kt + (size_t)bh * 1024 * 32;
    const unsigned short* vb = vv + (size_t)bh * 64 * 1024;

    // resident Q fragment for this wave's 16 rows (K = DK = 32 exactly)
    Frag qa;
    #pragma unroll
    for (int j = 0; j < 8; ++j)
        qa.u[j] = *(const unsigned int*)(qb + (size_t)(n0 + m16) * 32 + a_kidx(j, h16));

    v8f oacc[4];
    #pragma unroll
    for (int i = 0; i < 4; ++i) oacc[i] = {};
    float rowmax[8], rowsum[8];
    #pragma unroll
    for (int j = 0; j < 8; ++j) { rowmax[j] = -3.0e38f; rowsum[j] = 0.f; }

    // async tile copies: every thread issues exactly 1 b64 (K) + 1 b128 (V)
    auto issue_tile = [&](int buf, int m0) {
        int krow = t >> 3, c8 = t & 7;                    // 32 rows x 8B chunks
        async_ld_b64(lds_off(&Ks[buf][krow][c8 * 4]),
                     kb + (size_t)(m0 + krow) * 32 + c8 * 4);
        int e = t >> 2, c4 = t & 3;                        // 64 rows x 16B chunks
        async_ld_b128(lds_off(&Vs[buf][e][c4 * 8]),
                      vb + (size_t)e * 1024 + m0 + c4 * 8);
    };

    issue_tile(0, 0);

    for (int it = 0; it < 32; ++it) {
        const int m0 = it * 32;
        const int cur = it & 1;
        if (it + 1 < 32) {
            issue_tile(cur ^ 1, m0 + 32);
            wait_async_2();        // tile `it` done; tile `it+1` still in flight
        } else {
            wait_async_0();
        }
        __syncthreads();

        // K fragments from LDS (pairs contiguous in d)
        Frag kf0, kf1;
        #pragma unroll
        for (int j = 0; j < 8; ++j) {
            int kk = b_kidx(j, h16);
            kf0.u[j] = *(const unsigned int*)&Ks[cur][m16][kk];
            kf1.u[j] = *(const unsigned int*)&Ks[cur][16 + m16][kk];
        }
        v8f zc = {};
        v8f s0 = wmma_bf16(qa, kf0, zc);
        v8f s1 = wmma_bf16(qa, kf1, zc);

        // ---- online softmax over this 32-key block ----
        float bm[8];
        #pragma unroll
        for (int j = 0; j < 8; ++j) bm[j] = fmaxf(s0[j], s1[j]);
        #pragma unroll
        for (int off = 8; off >= 1; off >>= 1)
            #pragma unroll
            for (int j = 0; j < 8; ++j)
                bm[j] = fmaxf(bm[j], __shfl_xor(bm[j], off, 32));

        float corr[8];
        #pragma unroll
        for (int j = 0; j < 8; ++j) {
            float mn = fmaxf(rowmax[j], bm[j]);
            corr[j] = __expf(rowmax[j] - mn);
            rowmax[j] = mn;
            s0[j] = __expf(s0[j] - mn);
            s1[j] = __expf(s1[j] - mn);
        }
        float rs[8];
        #pragma unroll
        for (int j = 0; j < 8; ++j) rs[j] = s0[j] + s1[j];
        #pragma unroll
        for (int off = 8; off >= 1; off >>= 1)
            #pragma unroll
            for (int j = 0; j < 8; ++j)
                rs[j] += __shfl_xor(rs[j], off, 32);
        #pragma unroll
        for (int j = 0; j < 8; ++j) rowsum[j] = rowsum[j] * corr[j] + rs[j];
        #pragma unroll
        for (int ti = 0; ti < 4; ++ti)
            #pragma unroll
            for (int j = 0; j < 8; ++j) oacc[ti][j] *= corr[j];

        // restage P (C layout -> A layout) through per-wave LDS patch
        #pragma unroll
        for (int j = 0; j < 8; ++j) {
            int row = j + 8 * h16;
            Pl[wv][row][m16]      = f2bf(s0[j]);
            Pl[wv][row][16 + m16] = f2bf(s1[j]);
        }
        __builtin_amdgcn_wave_barrier();   // DS ops are in-order within a wave
        Frag pa;
        #pragma unroll
        for (int j = 0; j < 8; ++j)
            pa.u[j] = *(const unsigned int*)&Pl[wv][m16][a_kidx(j, h16)];
        __builtin_amdgcn_wave_barrier();

        // O += P * V   (4 tiles of 16 e-columns; pairs contiguous along m)
        #pragma unroll
        for (int ti = 0; ti < 4; ++ti) {
            Frag vf;
            #pragma unroll
            for (int j = 0; j < 8; ++j)
                vf.u[j] = *(const unsigned int*)&Vs[cur][ti * 16 + m16][b_kidx(j, h16)];
            oacc[ti] = wmma_bf16(pa, vf, oacc[ti]);
        }
        __syncthreads();   // protect buffer `cur` before it is refilled
    }

    // normalize, ReLU, store bf16 to [b][n][512] (channel = h*64 + e)
    #pragma unroll
    for (int ti = 0; ti < 4; ++ti)
        #pragma unroll
        for (int j = 0; j < 8; ++j) {
            int row = j + 8 * h16;
            int n = n0 + row;
            int e = ti * 16 + m16;
            float val = fmaxf(oacc[ti][j] / rowsum[j], 0.0f);
            at[((size_t)b * 1024 + n) * 512 + h * 64 + e] = f2bf(val);
        }
}

// ---------------------------------------------------------------------------
// Kernel 3: output projection  out[b][o][n] = BN(sum_c Wp[o][c]*at[b][n][c])
//   M = 384, K = 512, f32 output to d_out.
//   B operand is already bf16 in tile layout -> async global->LDS fill.
// ---------------------------------------------------------------------------
__global__ __launch_bounds__(256) void gemm_proj_kernel(
    const float* __restrict__ Wp, const unsigned short* __restrict__ Ain,
    const float* __restrict__ bng, const float* __restrict__ bnb,
    const float* __restrict__ bnm, const float* __restrict__ bnv,
    float* __restrict__ out)
{
    const int C = 512, N = 1024, M = 384;
    __shared__ __align__(16) unsigned short As[128][34];
    __shared__ __align__(16) unsigned short Bs[64][40];   // 80B pitch, 16B aligned

    const int b  = blockIdx.z;
    const int o0 = blockIdx.y * 128;
    const int n0 = blockIdx.x * 64;
    const int t  = threadIdx.x;
    const int wv = t >> 5, lane = t & 31;
    const int m16 = lane & 15, h16 = lane >> 4;
    const int om = (wv >> 1) * 32, on = (wv & 1) * 32;

    v8f acc[2][2];
    for (int i = 0; i < 2; ++i)
        for (int j = 0; j < 2; ++j) acc[i][j] = {};

    for (int kb = 0; kb < C; kb += 32) {
        // B tile: 64 rows x 64B, one async b128 per thread
        {
            int n = t >> 2, c4 = t & 3;
            async_ld_b128(lds_off(&Bs[n][c4 * 8]),
                          Ain + ((size_t)b * 1024 + n0 + n) * C + kb + c4 * 8);
        }
        // A tile: f32 weights -> bf16 (manual)
        #pragma unroll
        for (int i = 0; i < 16; ++i) {
            int idx = t + i * 256, r = idx >> 5, k = idx & 31;
            As[r][k] = f2bf(Wp[(size_t)(o0 + r) * C + kb + k]);
        }
        wait_async_0();
        __syncthreads();

        Frag a0, a1, bf0, bf1;
        #pragma unroll
        for (int j = 0; j < 8; ++j) {
            int ka = a_kidx(j, h16);
            int kbb = b_kidx(j, h16);
            a0.u[j]  = *(const unsigned int*)&As[om + m16][ka];
            a1.u[j]  = *(const unsigned int*)&As[om + 16 + m16][ka];
            bf0.u[j] = *(const unsigned int*)&Bs[on + m16][kbb];
            bf1.u[j] = *(const unsigned int*)&Bs[on + 16 + m16][kbb];
        }
        acc[0][0] = wmma_bf16(a0, bf0, acc[0][0]);
        acc[0][1] = wmma_bf16(a0, bf1, acc[0][1]);
        acc[1][0] = wmma_bf16(a1, bf0, acc[1][0]);
        acc[1][1] = wmma_bf16(a1, bf1, acc[1][1]);
        __syncthreads();
    }

    #pragma unroll
    for (int ti = 0; ti < 2; ++ti)
        #pragma unroll
        for (int tj = 0; tj < 2; ++tj)
            #pragma unroll
            for (int j = 0; j < 8; ++j) {
                int o = o0 + om + ti * 16 + j + 8 * h16;
                int n = n0 + on + tj * 16 + m16;
                float sc = bng[o] * rsqrtf(bnv[o] + EPS);
                float sh = bnb[o] - bnm[o] * sc;
                out[((size_t)b * M + o) * (size_t)N + n] = acc[ti][tj][j] * sc + sh;
            }
}

// ---------------------------------------------------------------------------
extern "C" void kernel_launch(void* const* d_in, const int* in_sizes, int n_in,
                              void* d_out, int out_size, void* d_ws, size_t ws_size,
                              hipStream_t stream)
{
    (void)in_sizes; (void)n_in; (void)out_size; (void)ws_size;

    const float* x      = (const float*)d_in[0];
    const float* w_q    = (const float*)d_in[1];
    const float* g_q    = (const float*)d_in[2];
    const float* b_q    = (const float*)d_in[3];
    const float* m_q    = (const float*)d_in[4];
    const float* var_q  = (const float*)d_in[5];
    const float* w_k    = (const float*)d_in[6];
    const float* g_k    = (const float*)d_in[7];
    const float* b_k    = (const float*)d_in[8];
    const float* m_k    = (const float*)d_in[9];
    const float* var_k  = (const float*)d_in[10];
    const float* w_v    = (const float*)d_in[11];
    const float* g_v    = (const float*)d_in[12];
    const float* b_v    = (const float*)d_in[13];
    const float* m_v    = (const float*)d_in[14];
    const float* var_v  = (const float*)d_in[15];
    const float* w_proj = (const float*)d_in[16];
    const float* g_p    = (const float*)d_in[17];
    const float* b_p    = (const float*)d_in[18];
    const float* m_p    = (const float*)d_in[19];
    const float* var_p  = (const float*)d_in[20];

    // workspace partition (bf16 elements)
    unsigned short* qt = (unsigned short*)d_ws;          // 8*8*1024*32 = 2M elem
    unsigned short* kt = qt + (size_t)8 * 8 * 1024 * 32;
    unsigned short* vb = kt + (size_t)8 * 8 * 1024 * 32; // 8*512*1024 = 4M elem
    unsigned short* at = vb + (size_t)8 * 512 * 1024;    // 8*1024*512 = 4M elem

    dim3 blk(256);
    gemm_qkv_kernel<<<dim3(16, 2, 8), blk, 0, stream>>>(
        w_q, x, g_q, b_q, m_q, var_q, qt, 256, 0);
    gemm_qkv_kernel<<<dim3(16, 2, 8), blk, 0, stream>>>(
        w_k, x, g_k, b_k, m_k, var_k, kt, 256, 0);
    gemm_qkv_kernel<<<dim3(16, 4, 8), blk, 0, stream>>>(
        w_v, x, g_v, b_v, m_v, var_v, vb, 512, 1);
    attn_kernel<<<dim3(8, 64), blk, 0, stream>>>(qt, kt, vb, at);
    gemm_proj_kernel<<<dim3(16, 3, 8), blk, 0, stream>>>(
        w_proj, at, g_p, b_p, m_p, var_p, (float*)d_out);
}
